// NCDNet_gen_36679020708618
// MI455X (gfx1250) — matrix operand order repaired and checked by reading, not compiled, not run on hardware
//
#include <hip/hip_runtime.h>
#include <stdint.h>

typedef __bf16 bf16;
typedef __attribute__((ext_vector_type(8)))  __bf16 bf16x8;
typedef __attribute__((ext_vector_type(16))) __bf16 bf16x16;
typedef __attribute__((ext_vector_type(8)))  float  f32x8;
typedef __attribute__((ext_vector_type(4)))  unsigned int u32x4_t;
typedef __attribute__((ext_vector_type(8)))  int i32x8_t;
typedef __attribute__((ext_vector_type(4)))  int i32x4_t;

#if defined(__has_builtin)
# if __has_builtin(__builtin_amdgcn_tensor_load_to_lds)
#  define HAVE_TDM 1
# endif
#endif
#ifndef HAVE_TDM
# define HAVE_TDM 0
#endif

#define TDM_KC 128   // k-chunk staged to LDS; 64 rows x 128 x 2B = 16KB per buffer

__device__ __forceinline__ float sigmoidf_(float x) { return 1.0f / (1.0f + __expf(-x)); }

__device__ __forceinline__ float urand_(uint32_t a, uint32_t b, uint32_t seed) {
    uint32_t h = a * 0x9E3779B9u + b * 0x85EBCA6Bu + seed * 0xC2B2AE35u;
    h ^= h >> 16; h *= 0x7FEB352Du; h ^= h >> 15; h *= 0x846CA68Bu; h ^= h >> 16;
    return (float)(h >> 8) * (1.0f / 16777216.0f);
}

__device__ __forceinline__ bf16x16 ldfrag_(const bf16* p0, const bf16* p1) {
    union { bf16x16 v; bf16x8 h[2]; } u;
    u.h[0] = *reinterpret_cast<const bf16x8*>(p0);
    u.h[1] = *reinterpret_cast<const bf16x8*>(p1);
    return u.v;
}

__device__ __forceinline__ f32x8 wmma_bf16_(bf16x16 a, bf16x16 b, f32x8 c) {
    return __builtin_amdgcn_wmma_f32_16x16x32_bf16(false, a, false, b, (short)0, c, false, false);
}

__device__ __forceinline__ void store_tile_(f32x8* acc, int row0, int col0, int al, int ah,
                                            const float* bias, float* outF, int ldof,
                                            bf16* outB, int ldob, int Nstore,
                                            int M, int N, int act) {
    #pragma unroll
    for (int t = 0; t < 4; ++t) {
        int col = col0 + t * 16 + al;
        #pragma unroll
        for (int r = 0; r < 8; ++r) {
            int row = row0 + ah * 8 + r;
            if (row >= M) continue;
            float v = 0.0f;
            if (col < N) {
                v = acc[t][r] + (bias ? bias[col] : 0.0f);
                if (act == 1)      v = sigmoidf_(v);
                else if (act == 2) v = fmaxf(v, 0.0f);
            }
            if (outF && col < N)      outF[(size_t)row * ldof + col] = v;
            if (outB && col < Nstore) outB[(size_t)row * ldob + col] = (bf16)v;
        }
    }
}

// ---------------------------------------------------------------------------
// Plain WMMA GEMM (global-load operands, software-pipelined fragments).
// ---------------------------------------------------------------------------
__global__ __launch_bounds__(256)
void wmma_gemm_bf16_k(const bf16* __restrict__ A, int lda,
                      const bf16* __restrict__ W, int ldw,
                      const float* __restrict__ bias,
                      float* __restrict__ outF, int ldof,
                      bf16* __restrict__ outB, int ldob, int Nstore,
                      int M, int N, int Kp, int act) {
    const int lane = threadIdx.x & 31;
    const int wave = threadIdx.x >> 5;
    const int row0 = (blockIdx.y * 8 + wave) * 16;
    if (row0 >= M) return;
    const int col0 = blockIdx.x * 64;
    const int al = lane & 15, ah = lane >> 4;

    int arow = row0 + al; if (arow >= M) arow = M - 1;
    const bf16* ap = A + (size_t)arow * lda + ah * 16;

    const bf16* bp[4];
    #pragma unroll
    for (int t = 0; t < 4; ++t) {
        int c = col0 + t * 16 + al;
        int cc = (c < N) ? c : (N - 1);
        bp[t] = W + (size_t)cc * ldw + ah * 16;
    }

    f32x8 acc[4] = {};
    bf16x16 a  = ldfrag_(ap, ap + 8);
    bf16x16 b0 = ldfrag_(bp[0], bp[0] + 8);
    bf16x16 b1 = ldfrag_(bp[1], bp[1] + 8);
    bf16x16 b2 = ldfrag_(bp[2], bp[2] + 8);
    bf16x16 b3 = ldfrag_(bp[3], bp[3] + 8);
    for (int kb = 32; kb < Kp; kb += 32) {
        bf16x16 an = ldfrag_(ap + kb, ap + kb + 8);
        bf16x16 n0 = ldfrag_(bp[0] + kb, bp[0] + kb + 8);
        bf16x16 n1 = ldfrag_(bp[1] + kb, bp[1] + kb + 8);
        bf16x16 n2 = ldfrag_(bp[2] + kb, bp[2] + kb + 8);
        bf16x16 n3 = ldfrag_(bp[3] + kb, bp[3] + kb + 8);
        acc[0] = wmma_bf16_(a, b0, acc[0]);
        acc[1] = wmma_bf16_(a, b1, acc[1]);
        acc[2] = wmma_bf16_(a, b2, acc[2]);
        acc[3] = wmma_bf16_(a, b3, acc[3]);
        a = an; b0 = n0; b1 = n1; b2 = n2; b3 = n3;
    }
    acc[0] = wmma_bf16_(a, b0, acc[0]);
    acc[1] = wmma_bf16_(a, b1, acc[1]);
    acc[2] = wmma_bf16_(a, b2, acc[2]);
    acc[3] = wmma_bf16_(a, b3, acc[3]);

    store_tile_(acc, row0, col0, al, ah, bias, outF, ldof, outB, ldob, Nstore, M, N, act);
}

// ---------------------------------------------------------------------------
// TDM-staged WMMA GEMM with ping-pong LDS buffers: Tensor Data Mover DMA of
// chunk c+1 overlaps compute on chunk c (wave0 waits tensorcnt<=1, in-order).
// ---------------------------------------------------------------------------
__device__ __forceinline__ void tdm_issue_(bf16* sh, const bf16* W, int ldw,
                                           int col0, int N, int kc0, int kw) {
#if HAVE_TDM
    if (threadIdx.x < 32) {
        uint64_t ga = (uint64_t)(uintptr_t)(W + (size_t)col0 * ldw + kc0);
        uint32_t lds = (uint32_t)(uintptr_t)sh;   // low 32 bits = LDS byte offset
        u32x4_t g0;
        g0[0] = 1u;                                            // count=1, user mode
        g0[1] = lds;                                           // lds_addr
        g0[2] = (uint32_t)(ga & 0xFFFFFFFFu);                  // global_addr[31:0]
        g0[3] = (uint32_t)((ga >> 32) & 0x01FFFFFFu) | (2u << 30); // addr[56:32] | type=2
        i32x8_t g1;
        g1[0] = (int)(1u << 16);                               // data_size=1 -> 2 bytes
        g1[1] = (int)(((uint32_t)ldw & 0xFFFFu) << 16);        // tensor_dim0[15:0]
        g1[2] = (int)((((uint32_t)ldw >> 16) & 0xFFFFu) |
                      (((uint32_t)N & 0xFFFFu) << 16));        // tensor_dim0[31:16] | tensor_dim1[15:0]
        g1[3] = (int)((((uint32_t)N >> 16) & 0xFFFFu) |
                      (((uint32_t)kw & 0xFFFFu) << 16));       // tensor_dim1[31:16] | tile_dim0
        g1[4] = (int)64;                                       // tile_dim1=64, tile_dim2=0
        g1[5] = ldw;                                           // tensor_dim0_stride[31:0]
        g1[6] = 0;
        g1[7] = 0;
        i32x4_t gz = {0, 0, 0, 0};
#if defined(__clang_major__) && (__clang_major__ >= 23)
        i32x8_t gz8 = {0, 0, 0, 0, 0, 0, 0, 0};
        __builtin_amdgcn_tensor_load_to_lds(g0, g1, gz, gz, gz8, 0);
#else
        __builtin_amdgcn_tensor_load_to_lds(g0, g1, gz, gz, 0);
#endif
    }
#else
    // cooperative fallback (also compiled by the host pass)
    const int nvec = (64 * kw) / 8;
    for (int i = threadIdx.x; i < nvec; i += 256) {
        int idx = i * 8;
        int r = idx / kw, c = idx % kw;
        int gr = col0 + r;
        bf16x8 v;
        if (gr < N) v = *reinterpret_cast<const bf16x8*>(W + (size_t)gr * ldw + kc0 + c);
        else        { bf16x8 z = {}; v = z; }
        *reinterpret_cast<bf16x8*>(sh + (size_t)r * kw + c) = v;
    }
#endif
}

__global__ __launch_bounds__(256)
void wmma_gemm_tdm_k(const bf16* __restrict__ A, int lda,
                     const bf16* __restrict__ W, int ldw,
                     const float* __restrict__ bias,
                     float* __restrict__ outF, int ldof,
                     bf16* __restrict__ outB, int ldob, int Nstore,
                     int M, int N, int Kp, int act) {
    __shared__ bf16 sh[2][64 * TDM_KC];
    const int lane = threadIdx.x & 31;
    const int wave = threadIdx.x >> 5;
    const int row0 = (blockIdx.y * 8 + wave) * 16;
    const int col0 = blockIdx.x * 64;
    const int al = lane & 15, ah = lane >> 4;

    int arow = row0 + al; if (arow >= M) arow = M - 1;   // no early return: barriers below
    const bf16* ap = A + (size_t)arow * lda + ah * 16;

    const int nch = (Kp + TDM_KC - 1) / TDM_KC;
    auto kwof = [&](int c) { int r = Kp - c * TDM_KC; return (r < TDM_KC) ? r : TDM_KC; };

    f32x8 acc[4] = {};
    tdm_issue_(sh[0], W, ldw, col0, N, 0, kwof(0));
    for (int c = 0; c < nch; ++c) {
        const int kc0 = c * TDM_KC;
        const int kw = kwof(c);
        if (c + 1 < nch)
            tdm_issue_(sh[(c + 1) & 1], W, ldw, col0, N, kc0 + TDM_KC, kwof(c + 1));
#if HAVE_TDM
        if (threadIdx.x < 32) {
            if (c + 1 < nch) __builtin_amdgcn_s_wait_tensorcnt(1); // chunk c resident, c+1 in flight
            else             __builtin_amdgcn_s_wait_tensorcnt(0);
        }
#endif
        __syncthreads();                      // tile visible to all waves
        const bf16* shc = sh[c & 1];
        for (int kb = 0; kb < kw; kb += 32) {
            bf16x16 a = ldfrag_(ap + kc0 + kb, ap + kc0 + kb + 8);
            #pragma unroll
            for (int t = 0; t < 4; ++t) {
                const bf16* bpp = shc + (size_t)(t * 16 + al) * kw + kb + ah * 16;
                bf16x16 b = ldfrag_(bpp, bpp + 8);
                acc[t] = wmma_bf16_(a, b, acc[t]);
            }
        }
        __syncthreads();                      // chunk consumed before its buffer is reused
    }
    if (row0 < M)
        store_tile_(acc, row0, col0, al, ah, bias, outF, ldof, outB, ldob, Nstore, M, N, act);
}

// ---------------------------------------------------------------------------
// Fused contrastive denominator: deno[i] = sum_j exp( (xn_i . yn_j) * invtemp )
// ---------------------------------------------------------------------------
__global__ __launch_bounds__(256)
void deno_wmma_k(const bf16* __restrict__ xn, const bf16* __restrict__ yn,
                 float* __restrict__ deno, int B, int Dp, float invtemp) {
    const int lane = threadIdx.x & 31;
    const int wave = threadIdx.x >> 5;
    const int row0 = (blockIdx.x * 8 + wave) * 16;
    if (row0 >= B) return;
    const int al = lane & 15, ah = lane >> 4;
    int arow = row0 + al; if (arow >= B) arow = B - 1;
    const bf16* ap = xn + (size_t)arow * Dp + ah * 16;

    float accl[8] = {0.f, 0.f, 0.f, 0.f, 0.f, 0.f, 0.f, 0.f};
    for (int jc = 0; jc < B; jc += 64) {
        const bf16* bp[4];
        #pragma unroll
        for (int t = 0; t < 4; ++t) {
            int jrow = jc + t * 16 + al; if (jrow >= B) jrow = B - 1;
            bp[t] = yn + (size_t)jrow * Dp + ah * 16;
        }
        f32x8 c[4] = {};
        bf16x16 a  = ldfrag_(ap, ap + 8);
        bf16x16 b0 = ldfrag_(bp[0], bp[0] + 8);
        bf16x16 b1 = ldfrag_(bp[1], bp[1] + 8);
        bf16x16 b2 = ldfrag_(bp[2], bp[2] + 8);
        bf16x16 b3 = ldfrag_(bp[3], bp[3] + 8);
        for (int kb = 32; kb < Dp; kb += 32) {
            bf16x16 an = ldfrag_(ap + kb, ap + kb + 8);
            bf16x16 n0 = ldfrag_(bp[0] + kb, bp[0] + kb + 8);
            bf16x16 n1 = ldfrag_(bp[1] + kb, bp[1] + kb + 8);
            bf16x16 n2 = ldfrag_(bp[2] + kb, bp[2] + kb + 8);
            bf16x16 n3 = ldfrag_(bp[3] + kb, bp[3] + kb + 8);
            c[0] = wmma_bf16_(a, b0, c[0]);
            c[1] = wmma_bf16_(a, b1, c[1]);
            c[2] = wmma_bf16_(a, b2, c[2]);
            c[3] = wmma_bf16_(a, b3, c[3]);
            a = an; b0 = n0; b1 = n1; b2 = n2; b3 = n3;
        }
        c[0] = wmma_bf16_(a, b0, c[0]);
        c[1] = wmma_bf16_(a, b1, c[1]);
        c[2] = wmma_bf16_(a, b2, c[2]);
        c[3] = wmma_bf16_(a, b3, c[3]);
        #pragma unroll
        for (int t = 0; t < 4; ++t)
            #pragma unroll
            for (int r = 0; r < 8; ++r)
                accl[r] += __expf(c[t][r] * invtemp);
    }
    #pragma unroll
    for (int r = 0; r < 8; ++r) {
        float v = accl[r];
        v += __shfl_xor(v, 8, 32);
        v += __shfl_xor(v, 4, 32);
        v += __shfl_xor(v, 2, 32);
        v += __shfl_xor(v, 1, 32);
        int row = row0 + ah * 8 + r;
        if (al == 0 && row < B) deno[row] = v;
    }
}

// ---------------------------------------------------------------------------
// Gather + dynamic mask + elementwise front-end (hashed Bernoulli mask).
// ---------------------------------------------------------------------------
__global__ __launch_bounds__(256)
void gather_mask_k(const int* __restrict__ stu_id, const int* __restrict__ ex_id,
                   const float* __restrict__ kp, const float* __restrict__ student_emb,
                   const float* __restrict__ k_diff_emb, const float* __restrict__ e_diff_emb,
                   const float* __restrict__ tok_stu, const float* __restrict__ tok_pro,
                   const float* __restrict__ stu_ratio, const float* __restrict__ exer_ratio,
                   bf16* __restrict__ mstu, bf16* __restrict__ xbf, float* __restrict__ mpro,
                   int B, int K, int Kp) {
    const int row = blockIdx.x;
    const int sid = stu_id[row], eid = ex_id[row];

    int n_p = (int)(1.0f * 0.05f * exer_ratio[eid]); if (n_p < 1) n_p = 1;
    float keep_p = (urand_(row, 0x1234u, 0xBEEFu) >= (float)n_p) ? 1.0f : 0.0f;
    float mp = e_diff_emb[eid] * keep_p + tok_pro[0] * (1.0f - keep_p);
    if (threadIdx.x == 0) mpro[row] = mp;
    const float ed = sigmoidf_(mp);

    int n_s = (int)((float)K * 0.05f * stu_ratio[sid]); if (n_s < 1) n_s = 1;
    const float p_s = (float)n_s / (float)K;

    for (int c = threadIdx.x; c < Kp; c += blockDim.x) {
        float msv = 0.0f, xv = 0.0f;
        if (c < K) {
            float keep = (urand_(row, c, 0xA5A5u) >= p_s) ? 1.0f : 0.0f;
            msv = student_emb[(size_t)sid * K + c] * keep + tok_stu[c] * (1.0f - keep);
            float stat = sigmoidf_(msv);
            float kd = sigmoidf_(k_diff_emb[(size_t)eid * K + c]);
            xv = ed * (stat - kd) * kp[(size_t)row * K + c];
        }
        mstu[(size_t)row * Kp + c] = (bf16)msv;
        xbf[(size_t)row * Kp + c]  = (bf16)xv;
    }
}

__global__ void wconv_k(const float* __restrict__ src, bf16* __restrict__ dst,
                        int K, int Kp, int doabs) {
    const int rowIdx = blockIdx.y;
    const int c = blockIdx.x * blockDim.x + threadIdx.x;
    if (c >= Kp) return;
    float v = 0.0f;
    if (c < K) { v = src[(size_t)rowIdx * K + c]; if (doabs) v = fabsf(v); }
    dst[(size_t)rowIdx * Kp + c] = (bf16)v;
}

__global__ void pro_l1_k(const float* __restrict__ mpro, const float* __restrict__ Wp1,
                         const float* __restrict__ bp1, bf16* __restrict__ out,
                         int HP, int HPp) {
    const int row = blockIdx.y;
    const int j = blockIdx.x * blockDim.x + threadIdx.x;
    if (j >= HPp) return;
    float v = 0.0f;
    if (j < HP) v = fmaxf(mpro[row] * Wp1[j] + bp1[j], 0.0f);
    out[(size_t)row * HPp + j] = (bf16)v;
}

__global__ __launch_bounds__(256)
void normalize_k(const float* __restrict__ z, const int* __restrict__ idx,
                 bf16* __restrict__ out, int D) {
    const int row = blockIdx.x;
    const int src = idx ? idx[row] : row;
    const float* zr = z + (size_t)src * D;
    float ss = 0.0f;
    for (int c = threadIdx.x; c < D; c += 256) { float v = zr[c]; ss += v * v; }
    for (int off = 16; off >= 1; off >>= 1) ss += __shfl_xor(ss, off, 32);
    __shared__ float red[8];
    const int wave = threadIdx.x >> 5, lane = threadIdx.x & 31;
    if (lane == 0) red[wave] = ss;
    __syncthreads();
    if (wave == 0) {
        float t = (lane < 8) ? red[lane] : 0.0f;
        for (int off = 4; off >= 1; off >>= 1) t += __shfl_xor(t, off, 32);
        if (lane == 0) red[0] = t;
    }
    __syncthreads();
    const float inv = 1.0f / fmaxf(sqrtf(red[0]), 1e-12f);
    for (int c = threadIdx.x; c < D; c += 256) out[(size_t)row * D + c] = (bf16)(zr[c] * inv);
}

__global__ __launch_bounds__(256)
void mole_k(const bf16* __restrict__ xn, const bf16* __restrict__ yn,
            float* __restrict__ mole, int B, int D, float invtemp) {
    const int wave = threadIdx.x >> 5, lane = threadIdx.x & 31;
    const int row = blockIdx.x * 8 + wave;
    if (row >= B) return;
    const bf16* xr = xn + (size_t)row * D;
    const bf16* yr = yn + (size_t)row * D;
    float d = 0.0f;
    for (int c = lane; c < D; c += 32) d += (float)xr[c] * (float)yr[c];
    for (int off = 16; off >= 1; off >>= 1) d += __shfl_xor(d, off, 32);
    if (lane == 0) mole[row] = __expf(d * invtemp);
}

__global__ __launch_bounds__(256)
void final_k(const bf16* __restrict__ h2, const float* __restrict__ W3,
             const float* __restrict__ b3, float* __restrict__ out, int B, int H2) {
    const int wave = threadIdx.x >> 5, lane = threadIdx.x & 31;
    const int row = blockIdx.x * 8 + wave;
    if (row >= B) return;
    const bf16* hr = h2 + (size_t)row * H2;
    float d = 0.0f;
    for (int c = lane; c < H2; c += 32) d += (float)hr[c] * fabsf(W3[c]);
    for (int off = 16; off >= 1; off >>= 1) d += __shfl_xor(d, off, 32);
    if (lane == 0) out[row] = sigmoidf_(d + b3[0]);
}

__global__ void zero2_k(float* acc) { if (threadIdx.x < 2) acc[threadIdx.x] = 0.0f; }

__global__ void loss_k(const float* __restrict__ mole, const float* __restrict__ deno,
                       float* __restrict__ acc, int B, int slot) {
    const int i = blockIdx.x * blockDim.x + threadIdx.x;
    if (i >= B) return;
    float v = -__logf(mole[i] / (deno[i] + 1e-8f) + 1e-8f);
    atomicAdd(&acc[slot], v);
}

__global__ void finalize_k(const float* __restrict__ acc, float* __restrict__ out, int B) {
    if (threadIdx.x == 0 && blockIdx.x == 0)
        out[0] = 2.0f * (acc[0] + acc[1]) / (float)B;
}

// ---------------------------------------------------------------------------
extern "C" void kernel_launch(void* const* d_in, const int* in_sizes, int n_in,
                              void* d_out, int out_size, void* d_ws, size_t ws_size,
                              hipStream_t stream) {
    (void)n_in; (void)out_size; (void)ws_size;
    const int*   stu_id      = (const int*)d_in[0];
    const int*   ex_id       = (const int*)d_in[1];
    const float* kp          = (const float*)d_in[2];
    const float* student_emb = (const float*)d_in[3];
    const float* k_diff_emb  = (const float*)d_in[4];
    const float* e_diff_emb  = (const float*)d_in[5];
    const float* tok_stu     = (const float*)d_in[6];
    const float* tok_pro     = (const float*)d_in[7];
    const float* usrprf      = (const float*)d_in[8];
    const float* itmprf      = (const float*)d_in[9];
    const float* stu_ratio   = (const float*)d_in[10];
    const float* exer_ratio  = (const float*)d_in[11];
    const float* W1 = (const float*)d_in[12]; const float* b1 = (const float*)d_in[13];
    const float* W2 = (const float*)d_in[14]; const float* b2 = (const float*)d_in[15];
    const float* W3 = (const float*)d_in[16]; const float* b3 = (const float*)d_in[17];
    const float* Ws1 = (const float*)d_in[18]; const float* bs1 = (const float*)d_in[19];
    const float* Ws2 = (const float*)d_in[20]; const float* bs2 = (const float*)d_in[21];
    const float* Wp1 = (const float*)d_in[22]; const float* bp1 = (const float*)d_in[23];
    const float* Wp2 = (const float*)d_in[24]; const float* bp2 = (const float*)d_in[25];

    const int B  = in_sizes[0];
    const int K  = in_sizes[6];    // 1024
    const int H1 = in_sizes[13];   // 512
    const int H2 = in_sizes[15];   // 256
    const int HS = in_sizes[19];   // 10512
    const int D  = in_sizes[21];   // 1024
    const int HP = in_sizes[23];   // 5512
    auto al32 = [](int v) { return (v + 31) & ~31; };
    const int Kp = al32(K), H1p = al32(H1), H2p = al32(H2);
    const int HSp = al32(HS), HPp = al32(HP), Dp = al32(D);
    const int Hbig = (HSp > HPp) ? HSp : HPp;

    size_t off = 0;
    char* base = (char*)d_ws;
    auto alloc = [&](size_t bytes) -> char* {
        char* p = base + off;
        off += (bytes + 255) & ~(size_t)255;
        return p;
    };
    bf16*  mstu  = (bf16*)alloc((size_t)B * Kp * 2);
    bf16*  xbf   = (bf16*)alloc((size_t)B * Kp * 2);
    float* mpro  = (float*)alloc((size_t)B * 4);
    bf16*  w1a   = (bf16*)alloc((size_t)H1 * Kp * 2);
    bf16*  w2a   = (bf16*)alloc((size_t)H2 * H1p * 2);
    bf16*  ws1b  = (bf16*)alloc((size_t)HS * Kp * 2);
    bf16*  ws2b  = (bf16*)alloc((size_t)D * HSp * 2);
    bf16*  wp2b  = (bf16*)alloc((size_t)D * HPp * 2);
    bf16*  h1    = (bf16*)alloc((size_t)B * H1p * 2);
    bf16*  h2    = (bf16*)alloc((size_t)B * H2p * 2);
    bf16*  hbig  = (bf16*)alloc((size_t)B * Hbig * 2);   // hs then hp
    float* zf    = (float*)alloc((size_t)B * Dp * 4);    // z_s then z_p
    bf16*  xns   = (bf16*)alloc((size_t)B * Dp * 2);
    bf16*  yns   = (bf16*)alloc((size_t)B * Dp * 2);
    bf16*  xnp   = (bf16*)alloc((size_t)B * Dp * 2);
    bf16*  ynp   = (bf16*)alloc((size_t)B * Dp * 2);
    float* moleS = (float*)alloc((size_t)B * 4);
    float* denoS = (float*)alloc((size_t)B * 4);
    float* moleP = (float*)alloc((size_t)B * 4);
    float* denoP = (float*)alloc((size_t)B * 4);
    float* acc   = (float*)alloc(2 * 4);

    float* outPred  = (float*)d_out;
    float* outRecon = (float*)d_out + B;

    const dim3 blk(256);
    auto gemmGrid = [&](int Nstore, int M) { return dim3((Nstore + 63) / 64, (M + 127) / 128); };

    zero2_k<<<1, 32, 0, stream>>>(acc);

    wconv_k<<<dim3((Kp + 255) / 256, H1), blk, 0, stream>>>(W1, w1a, K, Kp, 1);
    wconv_k<<<dim3((H1p + 255) / 256, H2), blk, 0, stream>>>(W2, w2a, H1, H1p, 1);
    wconv_k<<<dim3((Kp + 255) / 256, HS), blk, 0, stream>>>(Ws1, ws1b, K, Kp, 0);
    wconv_k<<<dim3((HSp + 255) / 256, D), blk, 0, stream>>>(Ws2, ws2b, HS, HSp, 0);
    wconv_k<<<dim3((HPp + 255) / 256, D), blk, 0, stream>>>(Wp2, wp2b, HP, HPp, 0);

    gather_mask_k<<<B, blk, 0, stream>>>(stu_id, ex_id, kp, student_emb, k_diff_emb,
                                         e_diff_emb, tok_stu, tok_pro, stu_ratio, exer_ratio,
                                         mstu, xbf, mpro, B, K, Kp);

    // prediction path (plain pipelined GEMM)
    wmma_gemm_bf16_k<<<gemmGrid(H1p, B), blk, 0, stream>>>(
        xbf, Kp, w1a, Kp, b1, nullptr, 0, h1, H1p, H1p, B, H1, Kp, 1);
    wmma_gemm_bf16_k<<<gemmGrid(H2p, B), blk, 0, stream>>>(
        h1, H1p, w2a, H1p, b2, nullptr, 0, h2, H2p, H2p, B, H2, H1p, 1);
    final_k<<<(B + 7) / 8, blk, 0, stream>>>(h2, W3, b3, outPred, B, H2);

    // SSL student side (TDM-staged, double-buffered weight tiles)
    wmma_gemm_tdm_k<<<gemmGrid(HSp, B), blk, 0, stream>>>(
        mstu, Kp, ws1b, Kp, bs1, nullptr, 0, hbig, HSp, HSp, B, HS, Kp, 2);
    wmma_gemm_tdm_k<<<gemmGrid(Dp, B), blk, 0, stream>>>(
        hbig, HSp, ws2b, HSp, bs2, zf, D, nullptr, 0, 0, B, D, HSp, 0);
    normalize_k<<<B, blk, 0, stream>>>(zf, nullptr, xns, D);
    normalize_k<<<B, blk, 0, stream>>>(usrprf, stu_id, yns, D);
    mole_k<<<(B + 7) / 8, blk, 0, stream>>>(xns, yns, moleS, B, Dp, 2.0f);
    deno_wmma_k<<<(B + 127) / 128, blk, 0, stream>>>(xns, yns, denoS, B, Dp, 2.0f);

    // SSL pro side (reuses hbig, zf)
    pro_l1_k<<<dim3((HPp + 255) / 256, B), blk, 0, stream>>>(mpro, Wp1, bp1, hbig, HP, HPp);
    wmma_gemm_tdm_k<<<gemmGrid(Dp, B), blk, 0, stream>>>(
        hbig, HPp, wp2b, HPp, bp2, zf, D, nullptr, 0, 0, B, D, HPp, 0);
    normalize_k<<<B, blk, 0, stream>>>(zf, nullptr, xnp, D);
    normalize_k<<<B, blk, 0, stream>>>(itmprf, ex_id, ynp, D);
    mole_k<<<(B + 7) / 8, blk, 0, stream>>>(xnp, ynp, moleP, B, Dp, 2.0f);
    deno_wmma_k<<<(B + 127) / 128, blk, 0, stream>>>(xnp, ynp, denoP, B, Dp, 2.0f);

    loss_k<<<(B + 255) / 256, blk, 0, stream>>>(moleS, denoS, acc, B, 0);
    loss_k<<<(B + 255) / 256, blk, 0, stream>>>(moleP, denoP, acc, B, 1);
    finalize_k<<<1, 32, 0, stream>>>(acc, outRecon, B);
}
